// GTA_Model_74887049773726
// MI455X (gfx1250) — matrix operand order repaired and checked by reading, not compiled, not run on hardware
//
#include <hip/hip_runtime.h>
#include <hip/hip_bf16.h>
#include <math.h>

typedef __attribute__((ext_vector_type(16))) _Float16 v16h;
typedef __attribute__((ext_vector_type(8)))  _Float16 v8h;
typedef __attribute__((ext_vector_type(8)))  float    v8f;
typedef __attribute__((ext_vector_type(4)))  float    v4f;
typedef __attribute__((ext_vector_type(4)))  unsigned int v4u;
typedef __attribute__((ext_vector_type(8)))  int      v8i;
typedef __attribute__((ext_vector_type(4)))  int      v4i;

#define C_NODES 32
#define T_LEN   768
#define D_DIM   100
#define H_HEADS 5
#define GCN_HID 60
#define FFN_DIM 64
#define NPRED   10
#define LHID    200
#define NL      3
#define LDH     128                    // padded f16 leading dim
#define ROWS    (C_NODES * T_LEN)      // 24576

static __device__ inline v16h cat16(v8h a, v8h b)
{
    union { v16h v; v8h h[2]; } u;
    u.h[0] = a; u.h[1] = b;
    return u.v;
}

// ---------------------------------------------------------------------------
// Tensor Data Mover: 2D f16 tile (tile_w x tile_h, row stride in elements)
// from global memory into LDS at byte offset lds_off. Issued per-wave.
// Descriptor layout per CDNA5 ISA ch.8 (count=1, type=2, data_size=2B).
// This toolchain exposes the 6-arg builtin:
//   (uint32x4 g0, int32x8 g1, int32x4 g2, int32x4 g3, int32x8, i32 cpol)
// ---------------------------------------------------------------------------
#if __has_builtin(__builtin_amdgcn_tensor_load_to_lds)
#define HAVE_TDM 1
static __device__ inline void tdm_load_2d(unsigned lds_off, const void* gaddr,
                                          unsigned tile_w, unsigned tile_h,
                                          unsigned row_stride)
{
    unsigned long long ga = (unsigned long long)(uintptr_t)gaddr;
    v4u g0;
    g0[0] = 1u;                                       // count=1, user mode
    g0[1] = lds_off;                                  // lds_addr (bytes)
    g0[2] = (unsigned)(ga & 0xffffffffu);             // global_addr lo
    g0[3] = (unsigned)((ga >> 32) & 0x01ffffffu) | (2u << 30);  // addr hi | type=2
    v8i g1;
    g1[0] = (int)(1u << 16);                          // data_size=1 (2 bytes)
    g1[1] = (int)((tile_w & 0xffffu) << 16);          // tensor_dim0[15:0]
    g1[2] = (int)(((tile_w >> 16) & 0xffffu) | ((tile_h & 0xffffu) << 16));
    g1[3] = (int)(((tile_h >> 16) & 0xffffu) | ((tile_w & 0xffffu) << 16)); // tile_dim0
    g1[4] = (int)(tile_h & 0xffffu);                  // tile_dim1 (tile_dim2=0)
    g1[5] = (int)row_stride;                          // tensor_dim0_stride lo
    g1[6] = 0;
    g1[7] = 0;
    v4i z4 = {0, 0, 0, 0};
    v8i z8 = {0, 0, 0, 0, 0, 0, 0, 0};
    __builtin_amdgcn_tensor_load_to_lds(g0, g1, z4, z4, z8, 0);
}
#else
#define HAVE_TDM 0
#endif

// ---------------------------------------------------------------------------
// WMMA GEMM, LDS-staged:  C[M,N] = op( A[M,K] * B[K,N] )
// ---------------------------------------------------------------------------
template <int A_F16>
__global__ __launch_bounds__(256) void wmma_gemm_kernel(
    const void* __restrict__ Ap, int lda,
    const float* __restrict__ B, int ldb,
    void* __restrict__ Cp, int out_mode, int ldc,
    const float* __restrict__ bias, int relu, int accum,
    int M, int N, int K, int Nw)
{
    __shared__ __align__(16) _Float16 sA[64 * 32];
    __shared__ __align__(16) _Float16 sB[32 * 64];

    const int tid   = threadIdx.x;
    const int lane  = tid & 31;
    const int wave  = tid >> 5;
    const int grp   = lane >> 4;
    const int lrow  = lane & 15;
    const int mw    = wave & 3;
    const int nw    = wave >> 2;
    const int mtile = blockIdx.x * 64;
    const int ntile = blockIdx.y * 64;

    const float*    Af = (const float*)Ap;
    const _Float16* Ah = (const _Float16*)Ap;

    v8f acc0 = {}, acc1 = {};

    for (int k0 = 0; k0 < K; k0 += 32) {
        if (k0 + 32 < K)   // global_prefetch_b8 on next B K-slab
            __builtin_prefetch((const char*)B + ((size_t)(k0 + 32) * ldb + ntile) * 4, 0, 3);

        {   // stage A slab: 64 rows x 32 halves
            const int row = tid >> 2;
            const int seg = (tid & 3) * 8;
            const int m   = mtile + row;
            const int ks  = k0 + seg;
            v8h val;
            if (A_F16) {
                if (m < M) val = *(const v8h*)(Ah + (size_t)m * lda + ks);
                else { v8h z = {}; val = z; }
            } else {
                if (m < M && ks + 8 <= K) {
                    v4f lo = *(const v4f*)(Af + (size_t)m * lda + ks);
                    v4f hi = *(const v4f*)(Af + (size_t)m * lda + ks + 4);
#pragma unroll
                    for (int i = 0; i < 4; ++i) { val[i] = (_Float16)lo[i]; val[4 + i] = (_Float16)hi[i]; }
                } else {
#pragma unroll
                    for (int i = 0; i < 8; ++i) {
                        int k = ks + i;
                        val[i] = (_Float16)((m < M && k < K) ? Af[(size_t)m * lda + k] : 0.0f);
                    }
                }
            }
            *(v8h*)(sA + row * 32 + seg) = val;
        }
        {   // stage B slab: 32 rows x 64 cols
            const int krow = tid >> 3;
            const int seg  = (tid & 7) * 8;
            const int k    = k0 + krow;
            const int n0   = ntile + seg;
            v8h val;
            if (k < K && n0 + 8 <= N) {
                v4f lo = *(const v4f*)(B + (size_t)k * ldb + n0);
                v4f hi = *(const v4f*)(B + (size_t)k * ldb + n0 + 4);
#pragma unroll
                for (int i = 0; i < 4; ++i) { val[i] = (_Float16)lo[i]; val[4 + i] = (_Float16)hi[i]; }
            } else {
#pragma unroll
                for (int i = 0; i < 8; ++i) {
                    int n = n0 + i;
                    val[i] = (_Float16)((k < K && n < N) ? B[(size_t)k * ldb + n] : 0.0f);
                }
            }
            *(v8h*)(sB + krow * 64 + seg) = val;
        }
        __syncthreads();

        {   // fragments + WMMA
            const int arow = mw * 16 + lrow;
            v16h af = cat16(*(const v8h*)(sA + arow * 32 + grp * 8),
                            *(const v8h*)(sA + arow * 32 + 16 + grp * 8));
            v16h bf0, bf1;
#pragma unroll
            for (int j = 0; j < 16; ++j) {
                const _Float16* bp = sB + (grp * 8 + j + (j >= 8 ? 8 : 0)) * 64 + nw * 32;
                bf0[j] = bp[lrow];
                bf1[j] = bp[16 + lrow];
            }
            acc0 = __builtin_amdgcn_wmma_f32_16x16x32_f16(false, af, false, bf0, (short)0, acc0, false, false);
            acc1 = __builtin_amdgcn_wmma_f32_16x16x32_f16(false, af, false, bf1, (short)0, acc1, false, false);
        }
        __syncthreads();
    }

    float*    Cf = (float*)Cp;
    _Float16* Ch = (_Float16*)Cp;
#pragma unroll
    for (int sub = 0; sub < 2; ++sub) {
        const v8f& acc = sub ? acc1 : acc0;
        const int n = ntile + nw * 32 + sub * 16 + lrow;
#pragma unroll
        for (int r = 0; r < 8; ++r) {
            const int m = mtile + mw * 16 + r + 8 * grp;
            if (m >= M || n >= Nw) continue;
            float v = acc[r];
            if (n < N) {
                if (accum && out_mode == 0) v += Cf[(size_t)m * ldc + n];
                if (bias)                   v += bias[n];
                if (relu && v < 0.0f)       v = 0.0f;
            } else {
                v = 0.0f;
            }
            if (out_mode == 0) {
                Cf[(size_t)m * ldc + n] = v;
            } else if (out_mode == 1) {
                Ch[(size_t)m * ldc + n] = (_Float16)v;
            } else {  // transposed f16: [c][n][t]
                const int c = m / T_LEN, t = m - c * T_LEN;
                Ch[(size_t)c * ldc * T_LEN + (size_t)n * T_LEN + t] = (_Float16)v;
            }
        }
    }
}

// ---------------------------------------------------------------------------
// Flash attention: block = 4 waves, 64 Q rows of one batch (= h*C + c).
// K tile (32 kv x 128 feat) and V tile (128 feat x 32 kv, from transposed Vt)
// are staged into LDS once per kv-step via the Tensor Data Mover and shared
// by all 4 waves (4x less global traffic than wave-private reads).
// ---------------------------------------------------------------------------
__global__ __launch_bounds__(128) void flash_attn_kernel(
    const _Float16* __restrict__ Q, const _Float16* __restrict__ Kb,
    const _Float16* __restrict__ Vt, _Float16* __restrict__ O, float scale)
{
    __shared__ __align__(16) _Float16 sK[32 * 128];
    __shared__ __align__(16) _Float16 sV[128 * 32];
    __shared__ __align__(16) _Float16 pls[4][16 * 32];

    const int lane = threadIdx.x & 31;
    const int wave = threadIdx.x >> 5;
    const int grp  = lane >> 4;
    const int lrow = lane & 15;
    const size_t base = (size_t)blockIdx.y * T_LEN * LDH;   // same for Q/K/Vt/O
    const int q0 = blockIdx.x * 64 + wave * 16;

    // Q fragments: two contiguous 8-half runs per feature chunk.
    v16h aq[4];
#pragma unroll
    for (int ch = 0; ch < 4; ++ch) {
        const _Float16* qp = Q + base + (size_t)(q0 + lrow) * LDH + ch * 32 + grp * 8;
        aq[ch] = cat16(*(const v8h*)qp, *(const v8h*)(qp + 16));
    }

    v8f o[8];
    float mrun[8], lrun[8];
#pragma unroll
    for (int i = 0; i < 8; ++i) { v8f z = {}; o[i] = z; mrun[i] = -1e30f; lrun[i] = 0.0f; }

    for (int kv0 = 0; kv0 < T_LEN; kv0 += 32) {
        __syncthreads();   // previous tile fully consumed
#if HAVE_TDM
        if (wave == 0) {
            tdm_load_2d((unsigned)(uintptr_t)&sK[0], Kb + base + (size_t)kv0 * LDH,
                        LDH, 32, LDH);            // 32 kv rows of 128 feats
            tdm_load_2d((unsigned)(uintptr_t)&sV[0], Vt + base + kv0,
                        32, LDH, T_LEN);          // 128 feat rows of 32 kvs
            __builtin_amdgcn_s_wait_tensorcnt(0);
        }
#else
        {   // cooperative fallback: 128 threads, vectorized loads
            const int row = threadIdx.x >> 2, seg = (threadIdx.x & 3) * 32;
#pragma unroll
            for (int i = 0; i < 4; ++i)
                *(v8h*)(sK + row * 128 + seg + i * 8) =
                    *(const v8h*)(Kb + base + (size_t)(kv0 + row) * LDH + seg + i * 8);
            const int vrow = threadIdx.x;   // feat row, 32 halves
#pragma unroll
            for (int i = 0; i < 4; ++i)
                *(v8h*)(sV + vrow * 32 + i * 8) =
                    *(const v8h*)(Vt + base + (size_t)vrow * T_LEN + kv0 + i * 8);
        }
#endif
        __syncthreads();

        // S = scale * Q K^T over 4 feature chunks, two 16-kv column tiles.
        v8f s0 = {}, s1 = {};
#pragma unroll
        for (int ch = 0; ch < 4; ++ch) {
            const _Float16* kp0 = sK + (size_t)lrow * LDH + ch * 32 + grp * 8;
            const _Float16* kp1 = kp0 + 16 * LDH;
            v16h bk0 = cat16(*(const v8h*)kp0, *(const v8h*)(kp0 + 16));
            v16h bk1 = cat16(*(const v8h*)kp1, *(const v8h*)(kp1 + 16));
            s0 = __builtin_amdgcn_wmma_f32_16x16x32_f16(false, aq[ch], false, bk0, (short)0, s0, false, false);
            s1 = __builtin_amdgcn_wmma_f32_16x16x32_f16(false, aq[ch], false, bk1, (short)0, s1, false, false);
        }

        // online softmax per q-row (row = r + 8*grp across 16 lanes)
#pragma unroll
        for (int r = 0; r < 8; ++r) {
            float x0 = s0[r] * scale, x1 = s1[r] * scale;
            float mx = fmaxf(x0, x1);
            mx = fmaxf(mx, __shfl_xor(mx, 1, 32));
            mx = fmaxf(mx, __shfl_xor(mx, 2, 32));
            mx = fmaxf(mx, __shfl_xor(mx, 4, 32));
            mx = fmaxf(mx, __shfl_xor(mx, 8, 32));
            float mnew = fmaxf(mrun[r], mx);
            float corr = __expf(mrun[r] - mnew);
            float p0 = __expf(x0 - mnew), p1 = __expf(x1 - mnew);
            float rs = p0 + p1;
            rs += __shfl_xor(rs, 1, 32);
            rs += __shfl_xor(rs, 2, 32);
            rs += __shfl_xor(rs, 4, 32);
            rs += __shfl_xor(rs, 8, 32);
            lrun[r] = lrun[r] * corr + rs;
            mrun[r] = mnew;
#pragma unroll
            for (int ft = 0; ft < 8; ++ft) o[ft][r] *= corr;
            const int row = r + 8 * grp;
            pls[wave][row * 32 + lrow]      = (_Float16)p0;
            pls[wave][row * 32 + 16 + lrow] = (_Float16)p1;
        }
        // wave-private P buffer: lockstep within the wave, no block barrier
        const _Float16* pp = pls[wave] + lrow * 32 + grp * 8;
        v16h ap = cat16(*(const v8h*)pp, *(const v8h*)(pp + 16));

        // O += P @ V from the shared LDS V tile.
#pragma unroll
        for (int ft = 0; ft < 8; ++ft) {
            const _Float16* vp = sV + (size_t)(ft * 16 + lrow) * 32 + grp * 8;
            v16h bv = cat16(*(const v8h*)vp, *(const v8h*)(vp + 16));
            o[ft] = __builtin_amdgcn_wmma_f32_16x16x32_f16(false, ap, false, bv, (short)0, o[ft], false, false);
        }
    }

#pragma unroll
    for (int r = 0; r < 8; ++r) {
        float inv = 1.0f / lrun[r];
        const int t = q0 + r + 8 * grp;
#pragma unroll
        for (int ft = 0; ft < 8; ++ft)
            O[base + (size_t)t * LDH + ft * 16 + lrow] = (_Float16)(o[ft][r] * inv);
    }
}

// ---------------------------------------------------------------------------
// Small helper kernels
// ---------------------------------------------------------------------------
__global__ __launch_bounds__(1024) void adj_norm_kernel(const float* __restrict__ A,
                                                        float* __restrict__ adjh)
{
    int i = threadIdx.x >> 5;
    int j = threadIdx.x & 31;
    float v = A[i * 32 + j] + (i == j ? 1.0f : 0.0f);
    float s = v;
    s += __shfl_xor(s, 1, 32);  s += __shfl_xor(s, 2, 32);
    s += __shfl_xor(s, 4, 32);  s += __shfl_xor(s, 8, 32);
    s += __shfl_xor(s, 16, 32);
    adjh[i * 32 + j] = v / s;
}

__global__ __launch_bounds__(32) void ln_residual_kernel(
    float* __restrict__ x, const float* __restrict__ r,
    const float* __restrict__ g, const float* __restrict__ b)
{
    int row = blockIdx.x, lane = threadIdx.x;
    float v[4]; float sum = 0.0f;
#pragma unroll
    for (int i = 0; i < 4; ++i) {
        int idx = lane + 32 * i;
        float val = 0.0f;
        if (idx < D_DIM) val = x[(size_t)row * D_DIM + idx] + r[(size_t)row * D_DIM + idx];
        v[i] = val; sum += val;
    }
    sum += __shfl_xor(sum, 1, 32); sum += __shfl_xor(sum, 2, 32);
    sum += __shfl_xor(sum, 4, 32); sum += __shfl_xor(sum, 8, 32);
    sum += __shfl_xor(sum, 16, 32);
    float mean = sum * (1.0f / D_DIM);
    float vs = 0.0f;
#pragma unroll
    for (int i = 0; i < 4; ++i) {
        int idx = lane + 32 * i;
        if (idx < D_DIM) { float d = v[i] - mean; vs += d * d; }
    }
    vs += __shfl_xor(vs, 1, 32); vs += __shfl_xor(vs, 2, 32);
    vs += __shfl_xor(vs, 4, 32); vs += __shfl_xor(vs, 8, 32);
    vs += __shfl_xor(vs, 16, 32);
    float inv = rsqrtf(vs * (1.0f / D_DIM) + 1e-5f);
#pragma unroll
    for (int i = 0; i < 4; ++i) {
        int idx = lane + 32 * i;
        if (idx < D_DIM)
            x[(size_t)row * D_DIM + idx] = (v[i] - mean) * inv * g[idx] + b[idx];
    }
}

__global__ void extract_last_kernel(const float* __restrict__ x, float* __restrict__ p)
{
    int idx = blockIdx.x * blockDim.x + threadIdx.x;
    if (idx >= C_NODES * D_DIM) return;
    int c = idx / D_DIM, d = idx % D_DIM;
    p[c * D_DIM + d] = x[((size_t)c * T_LEN + (T_LEN - 1)) * D_DIM + d];
}

__global__ void scatter_pred_kernel(const float* __restrict__ p, float* __restrict__ out, int step)
{
    int idx = blockIdx.x * blockDim.x + threadIdx.x;
    if (idx >= C_NODES * D_DIM) return;
    int c = idx / D_DIM, d = idx % D_DIM;
    out[(size_t)c * NPRED * D_DIM + step * D_DIM + d] = p[c * D_DIM + d];
}

// ---------------------------------------------------------------------------
static void gemm(hipStream_t s, const void* A, int a_f16, int lda,
                 const float* B, int ldb, void* Cp, int out_mode, int ldc,
                 const float* bias, int relu, int accum,
                 int M, int N, int K, int Nw)
{
    dim3 grid((M + 63) / 64, (Nw + 63) / 64);
    if (a_f16)
        wmma_gemm_kernel<1><<<grid, dim3(256), 0, s>>>(A, lda, B, ldb, Cp, out_mode, ldc,
                                                       bias, relu, accum, M, N, K, Nw);
    else
        wmma_gemm_kernel<0><<<grid, dim3(256), 0, s>>>(A, lda, B, ldb, Cp, out_mode, ldc,
                                                       bias, relu, accum, M, N, K, Nw);
}

extern "C" void kernel_launch(void* const* d_in, const int* in_sizes, int n_in,
                              void* d_out, int out_size, void* d_ws, size_t ws_size,
                              hipStream_t stream)
{
    (void)in_sizes; (void)n_in; (void)out_size; (void)ws_size;
    const float* X      = (const float*)d_in[0];
    const float* A      = (const float*)d_in[1];
    const float* gcn_w1 = (const float*)d_in[2];
    const float* gcn_b1 = (const float*)d_in[3];
    const float* gcn_w2 = (const float*)d_in[4];
    const float* gcn_b2 = (const float*)d_in[5];
    const float* enc_wq = (const float*)d_in[6];
    const float* enc_wk = (const float*)d_in[7];
    const float* enc_wv = (const float*)d_in[8];
    const float* enc_wm = (const float*)d_in[9];
    const float* enc_f1w = (const float*)d_in[10];
    const float* enc_f1b = (const float*)d_in[11];
    const float* enc_f2w = (const float*)d_in[12];
    const float* enc_f2b = (const float*)d_in[13];
    const float* ln1g = (const float*)d_in[14];
    const float* ln1b = (const float*)d_in[15];
    const float* ln2g = (const float*)d_in[16];
    const float* ln2b = (const float*)d_in[17];
    const float* lin1_w = (const float*)d_in[18];
    const float* lin1_b = (const float*)d_in[19];
    const float* lin2_w = (const float*)d_in[20];
    const float* lin2_b = (const float*)d_in[21];

    // ---- workspace carve-up -------------------------------------------------
    char* w = (char*)d_ws;
    float* adjh = (float*)w;                    w += 32 * 32 * 4;
    float* X1   = (float*)w;                    w += (size_t)ROWS * D_DIM * 4;
    float* T1   = (float*)w;                    w += (size_t)ROWS * D_DIM * 4;
    float* T2   = (float*)w;                    w += (size_t)ROWS * FFN_DIM * 4;
    const size_t fsz = (size_t)H_HEADS * ROWS * LDH;     // f16 elements per buffer
    _Float16* Qh  = (_Float16*)w;               w += fsz * 2;
    _Float16* Kh  = (_Float16*)w;               w += fsz * 2;
    _Float16* Vt  = (_Float16*)w;               w += fsz * 2;   // transposed V
    _Float16* AOh = (_Float16*)w;               w += fsz * 2;
    float* Pa   = (float*)w;                    w += C_NODES * D_DIM * 4;
    float* Pb   = (float*)w;                    w += C_NODES * D_DIM * 4;
    float* Ptmp = (float*)w;                    w += C_NODES * LHID * 4;

    (void)hipMemsetAsync(Qh, 0, fsz * 2 * 4, stream);   // zero pad features once

    // ---- GCN ---------------------------------------------------------------
    adj_norm_kernel<<<1, 1024, 0, stream>>>(A, adjh);
    gemm(stream, adjh, 0, 32, X, T_LEN * D_DIM, T1, 0, T_LEN * D_DIM,
         nullptr, 0, 0, 32, T_LEN * D_DIM, 32, T_LEN * D_DIM);
    gemm(stream, T1, 0, D_DIM, gcn_w1, GCN_HID, T2, 0, GCN_HID,
         gcn_b1, 1, 0, ROWS, GCN_HID, D_DIM, GCN_HID);
    gemm(stream, adjh, 0, 32, T2, T_LEN * GCN_HID, T1, 0, T_LEN * GCN_HID,
         nullptr, 0, 0, 32, T_LEN * GCN_HID, 32, T_LEN * GCN_HID);
    gemm(stream, T1, 0, GCN_HID, gcn_w2, D_DIM, X1, 0, D_DIM,
         gcn_b2, 0, 0, ROWS, D_DIM, GCN_HID, D_DIM);

    // ---- encoder layers ----------------------------------------------------
    for (int l = 0; l < NL; ++l) {
        for (int h = 0; h < H_HEADS; ++h) {
            const size_t woff = (size_t)(l * H_HEADS + h) * D_DIM * D_DIM;
            const size_t hoff = (size_t)h * ROWS * LDH;
            gemm(stream, X1, 0, D_DIM, enc_wq + woff, D_DIM, Qh + hoff, 1, LDH,
                 nullptr, 0, 0, ROWS, D_DIM, D_DIM, 112);
            gemm(stream, X1, 0, D_DIM, enc_wk + woff, D_DIM, Kh + hoff, 1, LDH,
                 nullptr, 0, 0, ROWS, D_DIM, D_DIM, 112);
            gemm(stream, X1, 0, D_DIM, enc_wv + woff, D_DIM, Vt + hoff, 2, LDH,
                 nullptr, 0, 0, ROWS, D_DIM, D_DIM, 112);
        }
        flash_attn_kernel<<<dim3(T_LEN / 64, H_HEADS * C_NODES), 128, 0, stream>>>(
            Qh, Kh, Vt, AOh, 0.1f /* 1/sqrt(D) */);
        for (int h = 0; h < H_HEADS; ++h)
            gemm(stream, AOh + (size_t)h * ROWS * LDH, 1, LDH,
                 enc_wm + (size_t)l * H_HEADS * D_DIM * D_DIM + (size_t)h * D_DIM * D_DIM,
                 D_DIM, T1, 0, D_DIM, nullptr, 0, (h > 0), ROWS, D_DIM, D_DIM, D_DIM);
        ln_residual_kernel<<<ROWS, 32, 0, stream>>>(X1, T1, ln1g + l * D_DIM, ln1b + l * D_DIM);
        gemm(stream, X1, 0, D_DIM, enc_f1w + (size_t)l * D_DIM * FFN_DIM, FFN_DIM,
             T2, 0, FFN_DIM, enc_f1b + l * FFN_DIM, 1, 0, ROWS, FFN_DIM, D_DIM, FFN_DIM);
        gemm(stream, T2, 0, FFN_DIM, enc_f2w + (size_t)l * FFN_DIM * D_DIM, D_DIM,
             T1, 0, D_DIM, enc_f2b + l * D_DIM, 0, 0, ROWS, D_DIM, FFN_DIM, D_DIM);
        ln_residual_kernel<<<ROWS, 32, 0, stream>>>(X1, T1, ln2g + l * D_DIM, ln2b + l * D_DIM);
    }

    // ---- autoregressive predictions ---------------------------------------
    extract_last_kernel<<<(C_NODES * D_DIM + 255) / 256, 256, 0, stream>>>(X1, Pa);
    float* cur = Pa; float* nxt = Pb;
    for (int p = 0; p < NPRED; ++p) {
        gemm(stream, cur, 0, D_DIM, lin1_w, LHID, Ptmp, 0, LHID,
             lin1_b, 0, 0, C_NODES, LHID, D_DIM, LHID);
        gemm(stream, Ptmp, 0, LHID, lin2_w, D_DIM, nxt, 0, D_DIM,
             lin2_b, 0, 0, C_NODES, D_DIM, LHID, D_DIM);
        scatter_pred_kernel<<<(C_NODES * D_DIM + 255) / 256, 256, 0, stream>>>(
            nxt, (float*)d_out, p);
        float* t = cur; cur = nxt; nxt = t;
    }
}
